// mutil_Attention_rf_24352464569480
// MI455X (gfx1250) — compile-verified, matrix operand order checked by reading, not run on hardware
//
#include <hip/hip_runtime.h>
#include <hip/hip_bf16.h>
#include <stdint.h>

// ---------------------------------------------------------------------------
// MI455X (gfx1250) fused multi-window attention block.
// All heavy math via v_wmma_f32_16x16x32_bf16 (f32 accum). Intermediates in
// bf16 NHWC so the 3x3-conv input (113 MB) is L2-resident (192 MB L2).
// BN folded into weights. Attention K tiles staged with
// global_load_async_to_lds_b128 (ASYNCcnt path).
// ---------------------------------------------------------------------------

typedef __bf16 bf16_t;
typedef __attribute__((ext_vector_type(16))) __bf16 v16bf;
typedef __attribute__((ext_vector_type(8)))  __bf16 v8bf;
typedef __attribute__((ext_vector_type(8)))  float  v8f;
typedef __attribute__((ext_vector_type(4)))  float  v4f;

#define NBATCH 4
#define CCH    384
#define HIMG   192
#define WIMG   192
#define HWIMG  36864          // 192*192
#define NPIX   147456         // 4*192*192

__device__ __forceinline__ v8f wmma_bf16(v16bf a, v16bf b, v8f c) {
  return __builtin_amdgcn_wmma_f32_16x16x32_bf16(false, a, false, b,
                                                 (short)0, c, false, false);
}

// 16-bit A/B fragment, documented wave32 layout: lane L (half = L>>4) holds
// K indices {half*8..+7} and {16+half*8..+7} -> two 16-byte loads.
__device__ __forceinline__ v16bf load_frag(const bf16_t* base, int half) {
  const v8bf lo = *reinterpret_cast<const v8bf*>(base + half * 8);
  const v8bf hi = *reinterpret_cast<const v8bf*>(base + 16 + half * 8);
  v16bf r;
#pragma unroll
  for (int i = 0; i < 8; ++i) { r[i] = lo[i]; r[8 + i] = hi[i]; }
  return r;
}

// Per-lane async 16B global->LDS copy (CDNA5, tracked by ASYNCcnt).
__device__ __forceinline__ void async_copy16(uint32_t lds_off,
                                             const bf16_t* gptr) {
  asm volatile("global_load_async_to_lds_b128 %0, %1, off"
               :: "v"(lds_off), "v"((uint64_t)(uintptr_t)gptr)
               : "memory");
}
__device__ __forceinline__ void wait_async0() {
  asm volatile("s_wait_asynccnt 0x0" ::: "memory");
}

// ---------------------------------------------------------------------------
// prep_qvk: fold input BN into qv_w/k_w -> bf16 Wqvk[1152][384] + f32 bias.
// ---------------------------------------------------------------------------
__global__ __launch_bounds__(384) void prep_qvk(
    const float* __restrict__ qv_w, const float* __restrict__ qv_b,
    const float* __restrict__ k_w,  const float* __restrict__ k_b,
    const float* __restrict__ g, const float* __restrict__ be,
    const float* __restrict__ mn, const float* __restrict__ vr,
    bf16_t* __restrict__ Wqvk, float* __restrict__ bqvk) {
  const int o = blockIdx.x;      // 0..1151
  const int c = threadIdx.x;     // 0..383
  const float s = g[c] * rsqrtf(vr[c] + 1e-5f);
  const float t = be[c] - mn[c] * s;
  const float w = (o < 768) ? qv_w[o * CCH + c] : k_w[(o - 768) * CCH + c];
  Wqvk[(size_t)o * CCH + c] = (bf16_t)(w * s);
  __shared__ float red[CCH];
  red[c] = w * t;
  __syncthreads();
  for (int st = 256; st > 0; st >>= 1) {
    if (c < st && c + st < CCH) red[c] += red[c + st];
    __syncthreads();
  }
  if (c == 0) {
    const float b0 = (o < 768) ? qv_b[o] : k_b[o - 768];
    bqvk[o] = b0 + red[0];
  }
}

// ---------------------------------------------------------------------------
// prep_oc: fold output BN into conv3x3 weights; OIHW -> [o][kh][kw][ci] bf16.
// ---------------------------------------------------------------------------
__global__ __launch_bounds__(384) void prep_oc(
    const float* __restrict__ oc_w, const float* __restrict__ g,
    const float* __restrict__ be, const float* __restrict__ mn,
    const float* __restrict__ vr, bf16_t* __restrict__ Woc,
    float* __restrict__ boc) {
  const int blk = blockIdx.x;            // o*9 + (kh*3+kw)
  const int o = blk / 9, khw = blk % 9;
  const int ci = threadIdx.x;
  const float s = g[o] * rsqrtf(vr[o] + 1e-5f);
  Woc[(size_t)blk * CCH + ci] =
      (bf16_t)(oc_w[((size_t)o * CCH + ci) * 9 + khw] * s);
  if (khw == 0 && ci == 0) boc[o] = be[o] - mn[o] * s;
}

// ---------------------------------------------------------------------------
// gemm_qvk: C[p][o] = sum_c X[c][p] * Wqvk[o][c] + b[o]
//   Block 64 pixels x 128 out-ch; 8 waves, wave tile 16x64 (4 wmma / A-frag).
//   Double-buffered LDS activation tile; staging packs 2 bf16 per dword.
// ---------------------------------------------------------------------------
__global__ __launch_bounds__(256) void gemm_qvk(
    const float* __restrict__ rgb, const float* __restrict__ fre,
    const bf16_t* __restrict__ Wqvk, const float* __restrict__ bqvk,
    bf16_t* __restrict__ Qb, bf16_t* __restrict__ Vb,
    bf16_t* __restrict__ Kb) {
  const int p0 = blockIdx.x * 64;
  const int o0 = blockIdx.y * 128;
  const float* __restrict__ X = (o0 >= 768) ? fre : rgb;
  const int b   = p0 / HWIMG;
  const int hw0 = p0 % HWIMG;
  const size_t xbase = (size_t)b * CCH * HWIMG + hw0;

  __shared__ bf16_t Xs[2][64][40];  // double-buffered 64px x 32ch tile

  const int tid = threadIdx.x;
  const int lane = tid & 31, wave = tid >> 5;
  const int wm = (wave & 3) * 16;      // wave M offset
  const int wn = (wave >> 2) * 64;     // wave N offset
  const int mloc = lane & 15, half = lane >> 4;

  // staging: each thread owns 2 adjacent channels x 4 pixels
  const int c2 = (tid & 15) * 2;       // channel pair base 0..30
  const int pp = (tid >> 4) * 4;       // pixel base 0..60

  v8f acc[4] = {{}, {}, {}, {}};
  v4f x0, x1;

  auto gload = [&](int cc) {
    const float* s0 = X + xbase + (size_t)(cc + c2) * HWIMG + pp;
    x0 = *reinterpret_cast<const v4f*>(s0);
    x1 = *reinterpret_cast<const v4f*>(s0 + HWIMG);
  };
  auto sstore = [&](int buf) {
#pragma unroll
    for (int j = 0; j < 4; ++j) {
      union { bf16_t h[2]; uint32_t u; } pk;
      pk.h[0] = (bf16_t)x0[j];
      pk.h[1] = (bf16_t)x1[j];
      *reinterpret_cast<uint32_t*>(&Xs[buf][pp + j][c2]) = pk.u;
    }
  };

  gload(0);
  sstore(0);
  for (int s = 0; s < 12; ++s) {
    if (s + 1 < 12) gload((s + 1) * 32);   // overlap next global loads
    __syncthreads();                        // buf[s&1] visible block-wide
    const v16bf a = load_frag(&Xs[s & 1][wm + mloc][0], half);
#pragma unroll
    for (int f = 0; f < 4; ++f) {
      const v16bf bw = load_frag(
          Wqvk + (size_t)(o0 + wn + f * 16 + mloc) * CCH + s * 32, half);
      acc[f] = wmma_bf16(a, bw, acc[f]);
    }
    if (s + 1 < 12) sstore((s + 1) & 1);   // write other buffer, no conflict
  }

  const float QSCALE = 0.08838834764831845f;  // 128^-0.5
#pragma unroll
  for (int f = 0; f < 4; ++f) {
    const int o = o0 + wn + f * 16 + mloc;    // C-frag column n = lane%16
    const float bias = bqvk[o];
#pragma unroll
    for (int rr = 0; rr < 8; ++rr) {
      const int m = rr + 8 * half;            // C-frag row
      const size_t p = (size_t)(p0 + wm + m);
      const float v = acc[f][rr] + bias;
      if (o < 384)      Qb[p * CCH + o]         = (bf16_t)(v * QSCALE);
      else if (o < 768) Vb[p * CCH + (o - 384)] = (bf16_t)v;
      else              Kb[p * CCH + (o - 768)] = (bf16_t)v;
    }
  }
}

// ---------------------------------------------------------------------------
// Window attention, d=128 per head group. 128 threads (4 waves), WPW windows
// per block. K staged via async global->LDS (token-major); V transposed to
// channel-major on store. S f32 in LDS, split-lane softmax, P bf16.
// ---------------------------------------------------------------------------
template <int WH, int WW, int WPW>
__global__ __launch_bounds__(128) void attn_kernel(
    const bf16_t* __restrict__ Qb, const bf16_t* __restrict__ Kb,
    const bf16_t* __restrict__ Vb, const float* __restrict__ rgb,
    bf16_t* __restrict__ Xc, int co) {
  constexpr int T  = WH * WW;
  constexpr int KT = (T < 32) ? 32 : T;   // padded K-dim for P@V
  constexpr int KP = 136;                 // K row stride (halves)
  constexpr int TP = KT + 8;              // Vt row stride (halves)
  constexpr int WAVES_PER_WIN = 4 / WPW;
  constexpr int QT = T / 16;
  constexpr int GH = HIMG / WH, GW = WIMG / WW;

  extern __shared__ char smem[];
  bf16_t* KsAll = (bf16_t*)smem;                       // WPW * T * KP
  bf16_t* VtAll = KsAll + WPW * T * KP;                // WPW * 128 * TP
  float*  Sall  = (float*)(VtAll + WPW * 128 * TP);    // 4 * 16 * T
  bf16_t* Pall  = (bf16_t*)(Sall + 4 * 16 * T);        // 4 * 16 * KT

  const int tid  = threadIdx.x;
  const int lane = tid & 31, wave = tid >> 5;
  const int slot = wave / WAVES_PER_WIN;
  const int sub  = wave % WAVES_PER_WIN;

  const int win = blockIdx.x * WPW + slot;
  const int b   = win / (GH * GW);
  const int rr_ = win % (GH * GW);
  const int h0  = (rr_ / GW) * WH;
  const int w0  = (rr_ % GW) * WW;

  bf16_t* Ks = KsAll + slot * T * KP;
  bf16_t* Vt = VtAll + slot * 128 * TP;
  float*  S  = Sall + wave * 16 * T;
  bf16_t* P  = Pall + wave * 16 * KT;

  auto pix = [&](int t) {
    return b * HWIMG + (h0 + t / WW) * WIMG + (w0 + t % WW);
  };

  // ---- stage K (async, natural) + V (transposed) into LDS ----
  {
    const int nthr = WAVES_PER_WIN * 32;
    const int ltid = sub * 32 + lane;
    for (int idx = ltid; idx < T * 16; idx += nthr) {
      const int t = idx >> 4, ch = (idx & 15) * 8;
      const size_t gb = (size_t)pix(t) * CCH + co + ch;
      const uint32_t loff = (uint32_t)((char*)(Ks + t * KP + ch) - smem);
      async_copy16(loff, Kb + gb);                    // K: 16B async to LDS
      const v8bf vv = *reinterpret_cast<const v8bf*>(Vb + gb);
#pragma unroll
      for (int j = 0; j < 8; ++j) Vt[(ch + j) * TP + t] = vv[j];
    }
    if constexpr (KT > T) {   // zero K-dim padding tokens of Vt
      for (int idx = ltid; idx < (KT - T) * 128; idx += nthr)
        Vt[(idx & 127) * TP + T + (idx >> 7)] = (bf16_t)0.0f;
    }
  }
  wait_async0();
  __syncthreads();

  const int mloc = lane & 15, half = lane >> 4;

  for (int qt = sub; qt < QT; qt += WAVES_PER_WIN) {
    // Q A-fragments (16 rows x d=128 -> 4 frags), softmax scale pre-applied
    v16bf qa[4];
    {
      const size_t qb = (size_t)pix(qt * 16 + mloc) * CCH + co;
#pragma unroll
      for (int kk = 0; kk < 4; ++kk) qa[kk] = load_frag(Qb + qb + kk * 32, half);
    }
    // ---- S = Qs @ K^T ----
    for (int j = 0; j < T / 16; ++j) {
      v8f acc = {};
#pragma unroll
      for (int kk = 0; kk < 4; ++kk) {
        const v16bf bk = load_frag(Ks + (j * 16 + mloc) * KP + kk * 32, half);
        acc = wmma_bf16(qa[kk], bk, acc);
      }
#pragma unroll
      for (int rr = 0; rr < 8; ++rr)
        S[(rr + 8 * half) * T + j * 16 + mloc] = acc[rr];
    }
    // ---- softmax rows: lane pair (L, L^16) splits columns ----
    {
      const int c0 = half * (KT / 2);
      float mx = -3.0e38f;
      for (int c = c0; c < c0 + KT / 2; ++c)
        if (c < T) mx = fmaxf(mx, S[mloc * T + c]);
      mx = fmaxf(mx, __shfl_xor(mx, 16, 32));
      float sum = 0.f;
      for (int c = c0; c < c0 + KT / 2; ++c)
        if (c < T) {
          const float e = __expf(S[mloc * T + c] - mx);
          S[mloc * T + c] = e;
          sum += e;
        }
      sum += __shfl_xor(sum, 16, 32);
      const float rs = 1.0f / sum;
      for (int c = c0; c < c0 + KT / 2; ++c)
        P[mloc * KT + c] = (c < T) ? (bf16_t)(S[mloc * T + c] * rs)
                                   : (bf16_t)0.0f;
    }
    // ---- O = P @ V, residual add, store NHWC bf16 ----
#pragma unroll 1
    for (int nt = 0; nt < 8; ++nt) {
      v8f acc = {};
      for (int kk = 0; kk < KT / 32; ++kk) {
        const v16bf pa = load_frag(P + mloc * KT + kk * 32, half);
        const v16bf vb = load_frag(Vt + (nt * 16 + mloc) * TP + kk * 32, half);
        acc = wmma_bf16(pa, vb, acc);
      }
      const int ch = co + nt * 16 + mloc;     // C-frag column
#pragma unroll
      for (int rr = 0; rr < 8; ++rr) {
        const int m = rr + 8 * half;
        const int p = pix(qt * 16 + m);
        const int hw = p - b * HWIMG;
        const float res = rgb[((size_t)b * CCH + ch) * HWIMG + hw];
        Xc[(size_t)p * CCH + ch] = (bf16_t)(acc[rr] + res);
      }
    }
  }
}

// ---------------------------------------------------------------------------
// conv3x3: implicit GEMM, K = 9*384, A-frags from L2-resident NHWC bf16.
// Block 64 pixels x 128 out-ch; wave tile 16x64 (4 wmma / A-frag).
// Fused bias + LeakyReLU(0.2), f32 NCHW output.
// ---------------------------------------------------------------------------
__global__ __launch_bounds__(256) void conv3x3(
    const bf16_t* __restrict__ Xc, const bf16_t* __restrict__ Woc,
    const float* __restrict__ boc, float* __restrict__ out) {
  const int p0 = blockIdx.x * 64;    // 64 pixels in one image row
  const int o0 = blockIdx.y * 128;
  const int tid = threadIdx.x;
  const int lane = tid & 31, wave = tid >> 5;
  const int wm = (wave & 3) * 16, wn = (wave >> 2) * 64;
  const int mloc = lane & 15, half = lane >> 4;

  const int p  = p0 + wm + mloc;     // lane's A-row pixel
  const int bb = p / HWIMG;
  const int hw = p % HWIMG;
  const int hh = hw / WIMG, wx = hw % WIMG;

  v8f acc[4] = {{}, {}, {}, {}};

  for (int kh = 0; kh < 3; ++kh) {
    const int hs = hh + kh - 1;
    const bool hok = (unsigned)hs < (unsigned)HIMG;
    for (int kw = 0; kw < 3; ++kw) {
      const int wsx = wx + kw - 1;
      const bool ok = hok && ((unsigned)wsx < (unsigned)WIMG);
      const size_t xrow = ((size_t)bb * HWIMG + hs * WIMG + wsx) * CCH;
      const size_t wbase =
          ((size_t)(o0 + wn + mloc) * 9 + kh * 3 + kw) * CCH;
#pragma unroll 1
      for (int cc = 0; cc < CCH; cc += 32) {
        v16bf a = {};
        if (ok) a = load_frag(Xc + xrow + cc, half);
#pragma unroll
        for (int f = 0; f < 4; ++f) {
          const v16bf bw =
              load_frag(Woc + wbase + (size_t)f * 16 * 9 * CCH + cc, half);
          acc[f] = wmma_bf16(a, bw, acc[f]);
        }
      }
    }
  }
#pragma unroll
  for (int f = 0; f < 4; ++f) {
    const int o = o0 + wn + f * 16 + mloc;
    const float bias = boc[o];
#pragma unroll
    for (int rr = 0; rr < 8; ++rr) {
      const int m = rr + 8 * half;
      const int pp = p0 + wm + m;
      const int b2 = pp / HWIMG, hw2 = pp % HWIMG;
      float v = acc[f][rr] + bias;
      v = (v >= 0.f) ? v : 0.2f * v;
      out[((size_t)b2 * CCH + o) * HWIMG + hw2] = v;
    }
  }
}

// ---------------------------------------------------------------------------
static size_t attn_smem(int T, int WPW) {
  const int KT = (T < 32) ? 32 : T;
  return (size_t)WPW * T * 136 * 2          // Ks
       + (size_t)WPW * 128 * (KT + 8) * 2   // Vt
       + (size_t)4 * 16 * T * 4             // S
       + (size_t)4 * 16 * KT * 2;           // P
}

extern "C" void kernel_launch(void* const* d_in, const int* in_sizes, int n_in,
                              void* d_out, int out_size, void* d_ws,
                              size_t ws_size, hipStream_t stream) {
  (void)in_sizes; (void)n_in; (void)out_size; (void)ws_size;
  const float* rgb  = (const float*)d_in[0];
  const float* fre  = (const float*)d_in[1];
  const float* ng   = (const float*)d_in[2];
  const float* nb   = (const float*)d_in[3];
  const float* nm   = (const float*)d_in[4];
  const float* nv   = (const float*)d_in[5];
  const float* qv_w = (const float*)d_in[6];
  const float* qv_b = (const float*)d_in[7];
  const float* k_w  = (const float*)d_in[8];
  const float* k_b  = (const float*)d_in[9];
  const float* oc_w = (const float*)d_in[10];
  const float* og   = (const float*)d_in[11];
  const float* ob   = (const float*)d_in[12];
  const float* om   = (const float*)d_in[13];
  const float* ov   = (const float*)d_in[14];
  float* out = (float*)d_out;

  char* ws = (char*)d_ws;
  auto carve = [&](size_t bytes) {
    void* pp = (void*)ws;
    ws += (bytes + 255) & ~(size_t)255;
    return pp;
  };
  bf16_t* Wqvk = (bf16_t*)carve((size_t)1152 * CCH * 2);
  bf16_t* Woc  = (bf16_t*)carve((size_t)CCH * 9 * CCH * 2);
  float*  bqvk = (float*)carve(1152 * 4);
  float*  boc  = (float*)carve(CCH * 4);
  bf16_t* Qb   = (bf16_t*)carve((size_t)NPIX * CCH * 2);
  bf16_t* Vb   = (bf16_t*)carve((size_t)NPIX * CCH * 2);
  bf16_t* Kb   = (bf16_t*)carve((size_t)NPIX * CCH * 2);
  bf16_t* Xc   = (bf16_t*)carve((size_t)NPIX * CCH * 2);

  prep_qvk<<<1152, 384, 0, stream>>>(qv_w, qv_b, k_w, k_b, ng, nb, nm, nv,
                                     Wqvk, bqvk);
  prep_oc<<<CCH * 9, 384, 0, stream>>>(oc_w, og, ob, om, ov, Woc, boc);

  gemm_qvk<<<dim3(NPIX / 64, 1152 / 128), 256, 0, stream>>>(
      rgb, fre, Wqvk, bqvk, Qb, Vb, Kb);

  // scale 0: 4x4 windows (T=16), 4 windows per block
  attn_kernel<4, 4, 4><<<(NBATCH * 48 * 48) / 4, 128, attn_smem(16, 4),
                         stream>>>(Qb, Kb, Vb, rgb, Xc, 0);
  // scale 1: 8x8 windows (T=64), 2 windows per block
  attn_kernel<8, 8, 2><<<(NBATCH * 24 * 24) / 2, 128, attn_smem(64, 2),
                         stream>>>(Qb, Kb, Vb, rgb, Xc, 128);
  // scale 2: 16x16 windows (T=256), 1 window per block (4 waves cooperate)
  attn_kernel<16, 16, 1><<<NBATCH * 12 * 12, 128, attn_smem(256, 1),
                           stream>>>(Qb, Kb, Vb, rgb, Xc, 256);

  conv3x3<<<dim3(NPIX / 64, CCH / 128), 256, 0, stream>>>(Xc, Woc, boc, out);
}
